// EnetGnn_69810398429320
// MI455X (gfx1250) — matrix operand order repaired and checked by reading, not compiled, not run on hardware
//
#include <hip/hip_runtime.h>
#include <math.h>

typedef __attribute__((ext_vector_type(16))) _Float16 v16h;
typedef __attribute__((ext_vector_type(8)))  _Float16 v8h;
typedef __attribute__((ext_vector_type(8)))  float    v8f;

#define NN   2
#define CC   128
#define C2   256
#define HH   64
#define WW   64
#define HW   4096
#define KNB  16   // neighbors

// ---------------------------------------------------------------------------
// h[n][p][c] = cnn[n][c][p]   (channels-last point features)
// ---------------------------------------------------------------------------
__global__ void k_build_h(const float* __restrict__ cnn, float* __restrict__ h) {
    int i = blockIdx.x * blockDim.x + threadIdx.x;     // over N*C*HW
    if (i >= NN * CC * HW) return;
    int n = i / (CC * HW);
    int r = i % (CC * HW);
    int c = r / HW;
    int p = r % HW;
    h[(n * HW + p) * CC + c] = cnn[i];
}

// ---------------------------------------------------------------------------
// KNN: per-thread top-16 smallest squared distances (tie-break: lower index),
// candidate features staged through LDS in tiles of 256 points.
// feat layout: [N, DIM, H, W]
// ---------------------------------------------------------------------------
template <int DIM>
__global__ void __launch_bounds__(256) k_knn(const float* __restrict__ feat,
                                             int* __restrict__ idx_out) {
    __shared__ float tile[256 * DIM];
    int n = blockIdx.y;
    int i = blockIdx.x * 256 + threadIdx.x;            // query point id
    const float* f = feat + (size_t)n * DIM * HW;

    float fi[DIM];
#pragma unroll
    for (int c = 0; c < DIM; ++c) fi[c] = f[c * HW + i];

    float bd[KNB];
    int   bi[KNB];
#pragma unroll
    for (int m = 0; m < KNB; ++m) { bd[m] = 3.4e38f; bi[m] = 0x7fffffff; }

    for (int tb = 0; tb < HW; tb += 256) {
        __syncthreads();
#pragma unroll
        for (int c = 0; c < DIM; ++c)
            tile[c * 256 + threadIdx.x] = f[c * HW + tb + threadIdx.x];
        __syncthreads();

        for (int jj = 0; jj < 256; ++jj) {
            int j = tb + jj;
            float d = 0.f;
#pragma unroll
            for (int c = 0; c < DIM; ++c) {
                float t = fi[c] - tile[c * 256 + jj];
                d += t * t;
            }
            if (d < bd[KNB - 1] || (d == bd[KNB - 1] && j < bi[KNB - 1])) {
                bd[KNB - 1] = d; bi[KNB - 1] = j;
#pragma unroll
                for (int m = KNB - 1; m > 0; --m) {
                    float d0 = bd[m - 1], d1 = bd[m];
                    int   i0 = bi[m - 1], i1 = bi[m];
                    bool sw = (d1 < d0) || (d1 == d0 && i1 < i0);
                    bd[m - 1] = sw ? d1 : d0; bd[m] = sw ? d0 : d1;
                    bi[m - 1] = sw ? i1 : i0; bi[m] = sw ? i0 : i1;
                }
            }
        }
    }
    int base = (n * HW + i) * KNB;
#pragma unroll
    for (int m = 0; m < KNB; ++m) idx_out[base + m] = bi[m];
}

// ---------------------------------------------------------------------------
// Pre-permute W [256 x 128] f32 into f16 WMMA B-fragments.
// Fragment (t,s): 32 lanes x 16 halves; element (lane,h):
//   N = t*16 + (lane&15),  K = s*32 + h + ((lane>>4)&1)*16
// ---------------------------------------------------------------------------
__global__ void k_wconv(const float* __restrict__ w, _Float16* __restrict__ wf) {
    int e = blockIdx.x * 256 + threadIdx.x;            // 32768 halves total
    int hh   = e & 15;
    int lane = (e >> 4) & 31;
    int s    = (e >> 9) & 7;
    int t    = (e >> 12) & 7;
    int Ncol = t * 16 + (lane & 15);
    int Krow = s * 32 + hh + ((lane >> 4) & 1) * 16;
    wf[e] = (_Float16)w[Krow * CC + Ncol];
}

// ---------------------------------------------------------------------------
// Edge MLP + neighbor-max, one point per block (128 threads = 4 waves).
// waves 0-1: rgb modality, waves 2-3: ir; each wave: 4 N-tiles x 8 K-steps
// of v_wmma_f32_16x16x32_f16.  M-axis of the 16x16 tile == neighbor axis,
// so max(axis=2) is 8 register maxes + shfl_xor(16).
// ---------------------------------------------------------------------------
__global__ void __launch_bounds__(128)
k_gnn_gemm(const float* __restrict__ h,
           const int* __restrict__ rgb_idx, const int* __restrict__ ir_idx,
           const _Float16* __restrict__ wf_rgb, const _Float16* __restrict__ wf_ir,
           const float* __restrict__ b_rgb, const float* __restrict__ b_ir,
           float* __restrict__ m_rgb, float* __restrict__ m_ir) {
    __shared__ _Float16 A_rgb[16 * 256];
    __shared__ _Float16 A_ir[16 * 256];

    int n = blockIdx.y;
    int p = blockIdx.x;
    int tid = threadIdx.x;
    int c = tid;                                        // channel 0..127

    const int* ri = rgb_idx + (n * HW + p) * KNB;
    const int* ii = ir_idx  + (n * HW + p) * KNB;
    const float* hb = h + (size_t)n * HW * CC;

    for (int r = 0; r < 16; ++r) {
        float rv = hb[(size_t)ri[r] * CC + c];
        float iv = hb[(size_t)ii[r] * CC + c];
        A_rgb[r * 256 + c]       = (_Float16)rv;
        A_rgb[r * 256 + 128 + c] = (_Float16)(rv - iv);
        A_ir[r * 256 + c]        = (_Float16)iv;
        A_ir[r * 256 + 128 + c]  = (_Float16)(iv - rv);
    }
    __syncthreads();

    int wave = tid >> 5;
    int lane = tid & 31;
    int hi   = (lane >> 4) & 1;
    int ln   = lane & 15;
    int mod  = wave >> 1;                               // 0=rgb 1=ir

    const _Float16* A  = mod ? A_ir : A_rgb;
    const v16h*     Wf = (const v16h*)(mod ? wf_ir : wf_rgb);
    const float* bias  = mod ? b_ir : b_rgb;
    float* mout = (mod ? m_ir : m_rgb) + ((size_t)n * HW + p) * CC;

    int t0 = (wave & 1) * 4;
#pragma unroll
    for (int t = t0; t < t0 + 4; ++t) {
        v8f acc = {};
#pragma unroll
        for (int s = 0; s < 8; ++s) {
            union { v16h v; v8h h8[2]; } a;
            // A k order used here (contiguous, same as B): halves 0..7 -> K = s*32 + h + hi*16
            //                  halves 8..15 -> K = s*32 + 8 + (h-8) + hi*16
            a.h8[0] = *(const v8h*)&A[ln * 256 + s * 32 + hi * 16];
            a.h8[1] = *(const v8h*)&A[ln * 256 + s * 32 + 8 + hi * 16];
            v16h b = Wf[(t * 8 + s) * 32 + lane];
            acc = __builtin_amdgcn_wmma_f32_16x16x32_f16(
                false, a.v, false, b, (short)0, acc, false, false);
        }
        float bv = bias[t * 16 + ln];
        float m = 0.0f;                                 // relu floor
#pragma unroll
        for (int r = 0; r < 8; ++r) m = fmaxf(m, acc[r] + bv);
        m = fmaxf(m, __shfl_xor(m, 16, 32));            // M and M+8 halves
        if (hi == 0) mout[t * 16 + ln] = m;
    }
}

// ---------------------------------------------------------------------------
// cat[n][j] = mean_p of m_rgb (j<128) / m_ir (j>=128)
// ---------------------------------------------------------------------------
__global__ void k_reduce_mean(const float* __restrict__ m_rgb,
                              const float* __restrict__ m_ir,
                              float* __restrict__ cat) {
    __shared__ float red[256];
    int n = blockIdx.y;
    int j = blockIdx.x;                                 // 0..255
    const float* src = (j < CC) ? (m_rgb + (size_t)n * HW * CC + j)
                                : (m_ir  + (size_t)n * HW * CC + (j - CC));
    float s = 0.f;
    for (int p = threadIdx.x; p < HW; p += 256) s += src[(size_t)p * CC];
    red[threadIdx.x] = s;
    __syncthreads();
    for (int off = 128; off > 0; off >>= 1) {
        if (threadIdx.x < off) red[threadIdx.x] += red[threadIdx.x + off];
        __syncthreads();
    }
    if (threadIdx.x == 0) cat[n * C2 + j] = red[0] * (1.0f / HW);
}

// ---------------------------------------------------------------------------
// SE gate: sigmoid(relu(cat @ w1 + b1) @ w2 + b2), one block per batch.
// ---------------------------------------------------------------------------
__global__ void __launch_bounds__(128)
k_se(const float* __restrict__ cat, const float* __restrict__ w1,
     const float* __restrict__ b1, const float* __restrict__ w2,
     const float* __restrict__ b2, float* __restrict__ gate) {
    __shared__ float hid[8];
    int n = blockIdx.x;
    int tid = threadIdx.x;
    if (tid < 8) {
        float s = b1[tid];
        for (int i = 0; i < C2; ++i) s += cat[n * C2 + i] * w1[i * 8 + tid];
        hid[tid] = fmaxf(s, 0.f);
    }
    __syncthreads();
    float g = b2[tid];
#pragma unroll
    for (int j = 0; j < 8; ++j) g += hid[j] * w2[j * CC + tid];
    gate[n * CC + tid] = 1.0f / (1.0f + expf(-g));
}

// ---------------------------------------------------------------------------
__global__ void k_scale(float* __restrict__ h, const float* __restrict__ gate) {
    int i = blockIdx.x * 256 + threadIdx.x;
    if (i >= NN * HW * CC) return;
    int n = i / (HW * CC);
    int c = i % CC;
    h[i] *= gate[n * CC + c];
}

__global__ void k_final(const float* __restrict__ h, const float* __restrict__ cnn,
                        const float* __restrict__ gamma, float* __restrict__ out) {
    int i = blockIdx.x * 256 + threadIdx.x;
    if (i >= NN * CC * HW) return;
    int n = i / (CC * HW);
    int r = i % (CC * HW);
    int c = r / HW;
    int p = r % HW;
    out[i] = fmaxf(gamma[0] * h[((size_t)n * HW + p) * CC + c] + cnn[i], 0.f);
}

// ---------------------------------------------------------------------------
extern "C" void kernel_launch(void* const* d_in, const int* in_sizes, int n_in,
                              void* d_out, int out_size, void* d_ws, size_t ws_size,
                              hipStream_t stream) {
    (void)in_sizes; (void)n_in; (void)out_size; (void)ws_size;
    const float* cnn   = (const float*)d_in[0];
    const float* rgb   = (const float*)d_in[1];
    const float* ir    = (const float*)d_in[2];
    const float* w_rgb = (const float*)d_in[3];
    const float* b_rgb = (const float*)d_in[4];
    const float* w_ir  = (const float*)d_in[5];
    const float* b_ir  = (const float*)d_in[6];
    const float* se1w  = (const float*)d_in[7];
    const float* se1b  = (const float*)d_in[8];
    const float* se2w  = (const float*)d_in[9];
    const float* se2b  = (const float*)d_in[10];
    const float* gamma = (const float*)d_in[11];
    // d_in[12] = gnn_iterations (2), d_in[13] = k (16): compile-time constants
    float* out = (float*)d_out;

    char* ws = (char*)d_ws;
    float*    h_buf   = (float*)(ws + 0);            // 4 MB
    float*    m_rgb   = (float*)(ws + 4194304);      // 4 MB
    float*    m_ir    = (float*)(ws + 8388608);      // 4 MB
    int*      rgb_idx = (int*)(ws + 12582912);       // 512 KB
    int*      ir_idx  = (int*)(ws + 13107200);       // 512 KB
    _Float16* wf_rgb  = (_Float16*)(ws + 13631488);  // 64 KB
    _Float16* wf_ir   = (_Float16*)(ws + 13697024);  // 64 KB
    float*    cat_buf = (float*)(ws + 13762560);     // 2 KB
    float*    gate    = (float*)(ws + 13764608);     // 1 KB

    const int TOT = NN * CC * HW;

    k_build_h<<<dim3((TOT + 255) / 256), dim3(256), 0, stream>>>(cnn, h_buf);

    k_knn<3><<<dim3(HW / 256, NN), dim3(256), 0, stream>>>(rgb, rgb_idx);
    k_knn<1><<<dim3(HW / 256, NN), dim3(256), 0, stream>>>(ir, ir_idx);

    k_wconv<<<dim3(32768 / 256), dim3(256), 0, stream>>>(w_rgb, wf_rgb);
    k_wconv<<<dim3(32768 / 256), dim3(256), 0, stream>>>(w_ir, wf_ir);

    for (int it = 0; it < 2; ++it) {
        k_gnn_gemm<<<dim3(HW, NN), dim3(128), 0, stream>>>(
            h_buf, rgb_idx, ir_idx, wf_rgb, wf_ir, b_rgb, b_ir, m_rgb, m_ir);
        k_reduce_mean<<<dim3(C2, NN), dim3(256), 0, stream>>>(m_rgb, m_ir, cat_buf);
        k_se<<<dim3(NN), dim3(128), 0, stream>>>(cat_buf, se1w, se1b, se2w, se2b, gate);
        k_scale<<<dim3((NN * HW * CC + 255) / 256), dim3(256), 0, stream>>>(h_buf, gate);
    }

    k_final<<<dim3((TOT + 255) / 256), dim3(256), 0, stream>>>(h_buf, cnn, gamma, out);
}
